// SG_knn_14628658610721
// MI455X (gfx1250) — matrix operand order repaired and checked by reading, not compiled, not run on hardware
//
#include <hip/hip_runtime.h>
#include <math.h>

#define BB 16
#define NN 1620
#define DD 64
#define CC 128
#define KNB 32
#define TOTAL_PTS (BB * NN)
#define NPAD 136           // LDS row stride in bf16 elems; 272B = 17*16B (b128-aligned, bank-rotating)
#define CNT_F ((float)(BB) * (float)(NN) * (float)(KNB))
#define EPSV 1e-5f

typedef __bf16 bf16_t;
typedef bf16_t v16bf __attribute__((ext_vector_type(16)));
typedef float  v8f   __attribute__((ext_vector_type(8)));
typedef unsigned uint_a __attribute__((may_alias));
typedef int v4i __attribute__((vector_size(4 * sizeof(int))));

struct __attribute__((may_alias)) U4 {
  unsigned x, y, z, w;
} __attribute__((aligned(16)));

union Frag16 {
  v16bf v;
  unsigned u[8];
  U4 q[2];
};

// ---- CDNA5 async global->LDS path (guarded; fallback = sync b128 copy) ----
#if defined(__has_builtin)
#if __has_builtin(__builtin_amdgcn_global_load_async_to_lds_b128)
#define HAVE_ASYNC 1
#endif
#endif
#ifndef HAVE_ASYNC
#define HAVE_ASYNC 0
#endif

#define ASG __attribute__((address_space(1)))
#define ASL __attribute__((address_space(3)))

__device__ inline void wait_async0() {
#if __has_builtin(__builtin_amdgcn_s_wait_asynccnt)
  __builtin_amdgcn_s_wait_asynccnt(0);
#else
  asm volatile("s_wait_asynccnt 0x0" ::: "memory");
#endif
  asm volatile("" ::: "memory");  // compiler barrier: LDS now holds async data
}

// ---------- bf16 helpers (round-to-nearest-even) ----------
__device__ inline bf16_t f2bf(float f) {
  unsigned u = __builtin_bit_cast(unsigned, f);
  unsigned r = u + 0x7FFFu + ((u >> 16) & 1u);
  unsigned short h = (unsigned short)(r >> 16);
  return __builtin_bit_cast(bf16_t, h);
}
__device__ inline float bf2f(bf16_t h) {
  unsigned short s = __builtin_bit_cast(unsigned short, h);
  unsigned u = ((unsigned)s) << 16;
  return __builtin_bit_cast(float, u);
}
__device__ inline v8f splat8(float x) {
  v8f z = {x, x, x, x, x, x, x, x};
  return z;
}

// ---------- Pass 0: transpose x -> (B,N,D) bf16, weights, wdiff, zero stats ----------
__global__ __launch_bounds__(256) void prep_kernel(
    const float* __restrict__ x, const float* __restrict__ w1,
    const float* __restrict__ w2, bf16_t* __restrict__ featsbf,
    bf16_t* __restrict__ w1bf, bf16_t* __restrict__ w2bf,
    float* __restrict__ stats, float* __restrict__ wdiff) {
  int gid = blockIdx.x * blockDim.x + threadIdx.x;
  int stride = gridDim.x * blockDim.x;
  const int total = BB * DD * NN;
  for (int i = gid; i < total; i += stride) {
    int n = i % NN;
    int t = i / NN;
    int c = t % DD;
    int b = t / DD;
    featsbf[((size_t)b * NN + n) * DD + c] = f2bf(x[i]);
  }
  if (gid < CC * CC) {
    w1bf[gid] = f2bf(w1[gid]);
    w2bf[gid] = f2bf(w2[gid]);
  }
  if (gid < CC * DD) {  // wdiff[o][c] = w1[o][64+c] - w1[o][c]  (exact f32)
    int o = gid >> 6, c = gid & 63;
    wdiff[gid] = w1[o * 128 + 64 + c] - w1[o * 128 + c];
  }
  if (gid < 1024) stats[gid] = 0.f;
}

// ---------- Pass 1: KNN (block = one batch chunk, coords staged in LDS) ----------
#define CHUNKS 8
#define PTSPC ((NN + CHUNKS - 1) / CHUNKS)

__global__ __launch_bounds__(256) void knn_kernel(const float* __restrict__ coords,
                                                  int* __restrict__ idxb) {
  __shared__ float cx[NN], cy[NN], cz[NN], cs[NN];
  int b = blockIdx.x / CHUNKS;
  int chunk = blockIdx.x % CHUNKS;
  const float* cb = coords + (size_t)b * NN * 3;
  for (int i = threadIdx.x; i < NN; i += 256) {
    float xx = cb[i * 3 + 0], yy = cb[i * 3 + 1], zz = cb[i * 3 + 2];
    cx[i] = xx; cy[i] = yy; cz[i] = zz;
    cs[i] = xx * xx + yy * yy + zz * zz;
  }
  __syncthreads();
  int n0 = chunk * PTSPC;
  int n1 = n0 + PTSPC; if (n1 > NN) n1 = NN;
  for (int n = n0 + threadIdx.x; n < n1; n += 256) {
    float bd[KNB]; int bi[KNB];
#pragma unroll
    for (int j = 0; j < KNB; ++j) { bd[j] = 3.4e38f; bi[j] = 0; }
    float mxv = 3.4e38f; int mxp = 0;
    float px = cx[n], py = cy[n], pz = cz[n], ps = cs[n];
    for (int m = 0; m < NN; ++m) {
      float d = ps + cs[m] - 2.f * (px * cx[m] + py * cy[m] + pz * cz[m]);
      if (d < mxv) {
        bd[mxp] = d; bi[mxp] = m;
        mxv = bd[0]; mxp = 0;
#pragma unroll
        for (int j = 1; j < KNB; ++j)
          if (bd[j] > mxv) { mxv = bd[j]; mxp = j; }
      }
    }
    int* op = idxb + ((size_t)b * NN + n) * KNB;
#pragma unroll
    for (int j = 0; j < KNB; ++j) op[j] = bi[j];
  }
}

// ---------- per-point machinery ----------
// Stage raw grouped rows (32 x 64 bf16) into padded per-wave LDS via async DMA.
__device__ inline void stage_grouped(bf16_t* mybuf, const bf16_t* fb,
                                     const int* nbr, int lane) {
  int sub = lane >> 3;  // 0..3 : row group
  int col = lane & 7;   // 16B chunk within 128B row
  int nb[8];
#pragma unroll
  for (int i = 0; i < 8; ++i) nb[i] = nbr[i * 4 + sub];
#pragma unroll
  for (int i = 0; i < 8; ++i) {
    int m = i * 4 + sub;
    const bf16_t* gsrc = fb + (size_t)nb[i] * DD + col * 8;
    bf16_t* ldst = mybuf + m * NPAD + col * 8;
#if HAVE_ASYNC
    __builtin_amdgcn_global_load_async_to_lds_b128(
        (ASG v4i*)(uintptr_t)gsrc, (ASL v4i*)(unsigned)(uintptr_t)ldst, 0, 0);
#else
    *(U4*)ldst = *(const U4*)gsrc;
#endif
  }
}

// Per-point rank-1 bias: bias[o] = sum_c ctr[c] * (w1[o][64+c]-w1[o][c]).
// Runs on VALU while the async A-tile DMA is in flight.
__device__ inline void compute_bias(const bf16_t* ctr, const float* wdiff,
                                    float* ctrbuf, float* biasbuf, int lane) {
  ctrbuf[lane] = bf2f(ctr[lane]);
  ctrbuf[lane + 32] = bf2f(ctr[lane + 32]);
  asm volatile("s_wait_dscnt 0x0" ::: "memory");
  const float4* wr0 = (const float4*)(wdiff + (lane + 0) * 64);
  const float4* wr1 = (const float4*)(wdiff + (lane + 32) * 64);
  const float4* wr2 = (const float4*)(wdiff + (lane + 64) * 64);
  const float4* wr3 = (const float4*)(wdiff + (lane + 96) * 64);
  const float4* c4p = (const float4*)ctrbuf;
  float s0 = 0.f, s1 = 0.f, s2 = 0.f, s3 = 0.f;
#pragma unroll
  for (int j = 0; j < 16; ++j) {
    float4 c4 = c4p[j];
    float4 w;
    w = wr0[j]; s0 += w.x * c4.x + w.y * c4.y + w.z * c4.z + w.w * c4.w;
    w = wr1[j]; s1 += w.x * c4.x + w.y * c4.y + w.z * c4.z + w.w * c4.w;
    w = wr2[j]; s2 += w.x * c4.x + w.y * c4.y + w.z * c4.z + w.w * c4.w;
    w = wr3[j]; s3 += w.x * c4.x + w.y * c4.y + w.z * c4.z + w.w * c4.w;
  }
  biasbuf[lane] = s0;
  biasbuf[lane + 32] = s1;
  biasbuf[lane + 64] = s2;
  biasbuf[lane + 96] = s3;
  asm volatile("s_wait_dscnt 0x0" ::: "memory");
}

// A fragments from LDS, 16-bit A 16x32 layout (ISA 7.12.2), NKS k-steps.
template <int NKS>
__device__ inline void load_afrags(const bf16_t* mybuf, int lane,
                                   Frag16 a[2][NKS]) {
  int mrow = lane & 15;
  int half = lane >> 4;
#pragma unroll
  for (int mt = 0; mt < 2; ++mt)
#pragma unroll
    for (int ks = 0; ks < NKS; ++ks) {
      const bf16_t* row = mybuf + (mt * 16 + mrow) * NPAD + ks * 32 + half * 8;
#pragma unroll
      for (int j = 0; j < 8; ++j) {
        int kk0 = ((j & 4) << 2) | ((j & 3) << 1);
        a[mt][ks].u[j] = *(const uint_a*)(row + kk0);
      }
    }
}

// B fragments: 16-bit B 32x16 layout; 2 x b128 global loads per k-step.
template <int NKS>
__device__ inline void load_bfrags(const bf16_t* wmat, int nt, int lane,
                                   Frag16 bfr[NKS]) {
  int ocol = lane & 15;
  int half = lane >> 4;
  const char* wb = (const char*)wmat + (size_t)(nt * 16 + ocol) * 256 + half * 16;
#pragma unroll
  for (int ks = 0; ks < NKS; ++ks) {
    const U4* q4 = (const U4*)(wb + ks * 64);
    bfr[ks].q[0] = q4[0];
    bfr[ks].q[1] = q4[2];
  }
}

template <int NKS>
__device__ inline void wmma_tile(const Frag16 a[2][NKS], const Frag16 bfr[NKS],
                                 v8f& acc0, v8f& acc1) {
#pragma unroll
  for (int ks = 0; ks < NKS; ++ks) {
    acc0 = __builtin_amdgcn_wmma_f32_16x16x32_bf16(false, a[0][ks].v, false,
                                                   bfr[ks].v, (short)0, acc0,
                                                   false, false);
    acc1 = __builtin_amdgcn_wmma_f32_16x16x32_bf16(false, a[1][ks].v, false,
                                                   bfr[ks].v, (short)0, acc1,
                                                   false, false);
  }
}

// BN1+ReLU; park y1 (bf16) into the (now free) per-wave LDS buffer.
__device__ inline void store_y1(bf16_t* mybuf, int nt, int lane, const v8f& acc0,
                                const v8f& acc1, float sc, float sh) {
  int ocol = lane & 15;
  int half = lane >> 4;
  int o = nt * 16 + ocol;
#pragma unroll
  for (int r = 0; r < 8; ++r) {
    float y0 = fmaxf(acc0[r] * sc + sh, 0.f);
    float y1 = fmaxf(acc1[r] * sc + sh, 0.f);
    mybuf[(r + 8 * half) * NPAD + o] = f2bf(y0);
    mybuf[(16 + r + 8 * half) * NPAD + o] = f2bf(y1);
  }
}

// ---------- Pass 2: GEMM1 (K=64 + rank-1 bias) + stats for BN1 ----------
__global__ __launch_bounds__(128) void gemm1_stats_kernel(
    const bf16_t* __restrict__ featsbf, const bf16_t* __restrict__ w1bf,
    const float* __restrict__ wdiff, const int* __restrict__ idxb,
    float* __restrict__ sum1, float* __restrict__ sq1) {
  __shared__ bf16_t buf[4][32 * NPAD];
  __shared__ __align__(16) float ctrls[4][64];
  __shared__ float biasls[4][128];
  int wave = threadIdx.x >> 5, lane = threadIdx.x & 31;
  bf16_t* mybuf = buf[wave];
  int gw = blockIdx.x * 4 + wave, nw = gridDim.x * 4;
  int ocol = lane & 15;
  float statS[8], statQ[8];
#pragma unroll
  for (int t = 0; t < 8; ++t) { statS[t] = 0.f; statQ[t] = 0.f; }
  for (int p = gw; p < TOTAL_PTS; p += nw) {
    int b = p / NN, n = p % NN;
    const bf16_t* fb = featsbf + (size_t)b * NN * DD;
    stage_grouped(mybuf, fb, idxb + (size_t)p * KNB, lane);
    compute_bias(fb + (size_t)n * DD, wdiff, ctrls[wave], biasls[wave], lane);
#if HAVE_ASYNC
    wait_async0();
#else
    asm volatile("s_wait_dscnt 0x0" ::: "memory");
#endif
    Frag16 a[2][2];
    load_afrags<2>(mybuf, lane, a);
    for (int nt = 0; nt < 8; ++nt) {
      Frag16 bfr[2];
      load_bfrags<2>(w1bf, nt, lane, bfr);
      float bs = biasls[wave][nt * 16 + ocol];
      v8f acc0 = splat8(bs), acc1 = splat8(bs);
      wmma_tile<2>(a, bfr, acc0, acc1);
#pragma unroll
      for (int r = 0; r < 8; ++r) {
        float v0 = acc0[r], v1 = acc1[r];
        statS[nt] += v0 + v1;
        statQ[nt] += v0 * v0 + v1 * v1;
      }
    }
  }
#pragma unroll
  for (int nt = 0; nt < 8; ++nt) {
    float s = statS[nt] + __shfl_xor(statS[nt], 16, 32);
    float q = statQ[nt] + __shfl_xor(statQ[nt], 16, 32);
    if ((lane >> 4) == 0) {
      atomicAdd(&sum1[nt * 16 + (lane & 15)], s);
      atomicAdd(&sq1[nt * 16 + (lane & 15)], q);
    }
  }
}

// ---------- stats -> (scale, shift) ----------
__global__ __launch_bounds__(128) void finalize_kernel(
    const float* __restrict__ sum, const float* __restrict__ sq,
    const float* __restrict__ gamma, const float* __restrict__ beta,
    float* __restrict__ scale, float* __restrict__ shift) {
  int o = threadIdx.x;
  float mean = sum[o] / CNT_F;
  float var = sq[o] / CNT_F - mean * mean;
  float sc = gamma[o] * rsqrtf(var + EPSV);
  scale[o] = sc;
  shift[o] = beta[o] - mean * sc;
}

// ---------- Pass 4: recompute GEMM1, BN1+ReLU, GEMM2 + stats for BN2 ----------
__global__ __launch_bounds__(128) void gemm2_stats_kernel(
    const bf16_t* __restrict__ featsbf, const bf16_t* __restrict__ w1bf,
    const bf16_t* __restrict__ w2bf, const float* __restrict__ wdiff,
    const int* __restrict__ idxb, const float* __restrict__ scale1,
    const float* __restrict__ shift1, float* __restrict__ sum2,
    float* __restrict__ sq2) {
  __shared__ bf16_t buf[4][32 * NPAD];
  __shared__ __align__(16) float ctrls[4][64];
  __shared__ float biasls[4][128];
  int wave = threadIdx.x >> 5, lane = threadIdx.x & 31;
  bf16_t* mybuf = buf[wave];
  int gw = blockIdx.x * 4 + wave, nw = gridDim.x * 4;
  int ocol = lane & 15;
  float sc1[8], sh1[8], statS[8], statQ[8];
#pragma unroll
  for (int t = 0; t < 8; ++t) {
    sc1[t] = scale1[t * 16 + ocol];
    sh1[t] = shift1[t * 16 + ocol];
    statS[t] = 0.f; statQ[t] = 0.f;
  }
  for (int p = gw; p < TOTAL_PTS; p += nw) {
    int b = p / NN, n = p % NN;
    const bf16_t* fb = featsbf + (size_t)b * NN * DD;
    stage_grouped(mybuf, fb, idxb + (size_t)p * KNB, lane);
    compute_bias(fb + (size_t)n * DD, wdiff, ctrls[wave], biasls[wave], lane);
#if HAVE_ASYNC
    wait_async0();
#else
    asm volatile("s_wait_dscnt 0x0" ::: "memory");
#endif
    Frag16 a[2][2];
    load_afrags<2>(mybuf, lane, a);
    for (int nt = 0; nt < 8; ++nt) {
      Frag16 bfr[2];
      load_bfrags<2>(w1bf, nt, lane, bfr);
      float bs = biasls[wave][nt * 16 + ocol];
      v8f acc0 = splat8(bs), acc1 = splat8(bs);
      wmma_tile<2>(a, bfr, acc0, acc1);
      store_y1(mybuf, nt, lane, acc0, acc1, sc1[nt], sh1[nt]);
    }
    asm volatile("s_wait_dscnt 0x0" ::: "memory");
    Frag16 a2[2][4];
    load_afrags<4>(mybuf, lane, a2);
    for (int nt = 0; nt < 8; ++nt) {
      Frag16 bfr[4];
      load_bfrags<4>(w2bf, nt, lane, bfr);
      v8f acc0 = splat8(0.f), acc1 = splat8(0.f);
      wmma_tile<4>(a2, bfr, acc0, acc1);
#pragma unroll
      for (int r = 0; r < 8; ++r) {
        float v0 = acc0[r], v1 = acc1[r];
        statS[nt] += v0 + v1;
        statQ[nt] += v0 * v0 + v1 * v1;
      }
    }
  }
#pragma unroll
  for (int nt = 0; nt < 8; ++nt) {
    float s = statS[nt] + __shfl_xor(statS[nt], 16, 32);
    float q = statQ[nt] + __shfl_xor(statQ[nt], 16, 32);
    if ((lane >> 4) == 0) {
      atomicAdd(&sum2[nt * 16 + (lane & 15)], s);
      atomicAdd(&sq2[nt * 16 + (lane & 15)], q);
    }
  }
}

// ---------- Pass 6: full recompute, BN2+ReLU, k-max, write (B,C,N) ----------
__global__ __launch_bounds__(128) void final_kernel(
    const bf16_t* __restrict__ featsbf, const bf16_t* __restrict__ w1bf,
    const bf16_t* __restrict__ w2bf, const float* __restrict__ wdiff,
    const int* __restrict__ idxb, const float* __restrict__ scale1,
    const float* __restrict__ shift1, const float* __restrict__ scale2,
    const float* __restrict__ shift2, float* __restrict__ out) {
  __shared__ bf16_t buf[4][32 * NPAD];
  __shared__ __align__(16) float ctrls[4][64];
  __shared__ float biasls[4][128];
  int wave = threadIdx.x >> 5, lane = threadIdx.x & 31;
  bf16_t* mybuf = buf[wave];
  int gw = blockIdx.x * 4 + wave, nw = gridDim.x * 4;
  int ocol = lane & 15, half = lane >> 4;
  float sc1[8], sh1[8], sc2[8], sh2[8];
#pragma unroll
  for (int t = 0; t < 8; ++t) {
    sc1[t] = scale1[t * 16 + ocol];
    sh1[t] = shift1[t * 16 + ocol];
    sc2[t] = scale2[t * 16 + ocol];
    sh2[t] = shift2[t * 16 + ocol];
  }
  for (int p = gw; p < TOTAL_PTS; p += nw) {
    int b = p / NN, n = p % NN;
    const bf16_t* fb = featsbf + (size_t)b * NN * DD;
    stage_grouped(mybuf, fb, idxb + (size_t)p * KNB, lane);
    compute_bias(fb + (size_t)n * DD, wdiff, ctrls[wave], biasls[wave], lane);
#if HAVE_ASYNC
    wait_async0();
#else
    asm volatile("s_wait_dscnt 0x0" ::: "memory");
#endif
    Frag16 a[2][2];
    load_afrags<2>(mybuf, lane, a);
    for (int nt = 0; nt < 8; ++nt) {
      Frag16 bfr[2];
      load_bfrags<2>(w1bf, nt, lane, bfr);
      float bs = biasls[wave][nt * 16 + ocol];
      v8f acc0 = splat8(bs), acc1 = splat8(bs);
      wmma_tile<2>(a, bfr, acc0, acc1);
      store_y1(mybuf, nt, lane, acc0, acc1, sc1[nt], sh1[nt]);
    }
    asm volatile("s_wait_dscnt 0x0" ::: "memory");
    Frag16 a2[2][4];
    load_afrags<4>(mybuf, lane, a2);
    for (int nt = 0; nt < 8; ++nt) {
      Frag16 bfr[4];
      load_bfrags<4>(w2bf, nt, lane, bfr);
      v8f acc0 = splat8(0.f), acc1 = splat8(0.f);
      wmma_tile<4>(a2, bfr, acc0, acc1);
      float sc = sc2[nt], sh = sh2[nt];
      float mx = 0.f;  // ReLU output >= 0, so 0 is max-identity
#pragma unroll
      for (int r = 0; r < 8; ++r) {
        mx = fmaxf(mx, fmaxf(acc0[r] * sc + sh, 0.f));
        mx = fmaxf(mx, fmaxf(acc1[r] * sc + sh, 0.f));
      }
      mx = fmaxf(mx, __shfl_xor(mx, 16, 32));
      if (half == 0)
        out[((size_t)b * CC + (nt * 16 + ocol)) * NN + n] = mx;
    }
  }
}

// ---------- host-side orchestration ----------
extern "C" void kernel_launch(void* const* d_in, const int* in_sizes, int n_in,
                              void* d_out, int out_size, void* d_ws,
                              size_t ws_size, hipStream_t stream) {
  (void)in_sizes; (void)n_in; (void)out_size; (void)ws_size;
  const float* x = (const float*)d_in[0];
  const float* coords = (const float*)d_in[1];
  const float* w1 = (const float*)d_in[2];
  const float* g1 = (const float*)d_in[3];
  const float* b1 = (const float*)d_in[4];
  const float* w2 = (const float*)d_in[5];
  const float* g2 = (const float*)d_in[6];
  const float* b2 = (const float*)d_in[7];
  float* out = (float*)d_out;

  char* ws = (char*)d_ws;
  size_t off = 0;
  bf16_t* featsbf = (bf16_t*)(ws + off); off += (size_t)BB * NN * DD * 2;
  bf16_t* w1bf = (bf16_t*)(ws + off);    off += (size_t)CC * CC * 2;
  bf16_t* w2bf = (bf16_t*)(ws + off);    off += (size_t)CC * CC * 2;
  int* idxb = (int*)(ws + off);          off += (size_t)BB * NN * KNB * 4;
  float* stats = (float*)(ws + off);     off += 1024 * 4;
  float* wdiff = (float*)(ws + off);     off += (size_t)CC * DD * 4;
  float* sum1 = stats + 0,   *sq1 = stats + 128;
  float* scale1 = stats + 256, *shift1 = stats + 384;
  float* sum2 = stats + 512, *sq2 = stats + 640;
  float* scale2 = stats + 768, *shift2 = stats + 896;

  prep_kernel<<<1024, 256, 0, stream>>>(x, w1, w2, featsbf, w1bf, w2bf, stats,
                                        wdiff);
  knn_kernel<<<BB * CHUNKS, 256, 0, stream>>>(coords, idxb);
  gemm1_stats_kernel<<<512, 128, 0, stream>>>(featsbf, w1bf, wdiff, idxb, sum1,
                                              sq1);
  finalize_kernel<<<1, 128, 0, stream>>>(sum1, sq1, g1, b1, scale1, shift1);
  gemm2_stats_kernel<<<512, 128, 0, stream>>>(featsbf, w1bf, w2bf, wdiff, idxb,
                                              scale1, shift1, sum2, sq2);
  finalize_kernel<<<1, 128, 0, stream>>>(sum2, sq2, g2, b2, scale2, shift2);
  final_kernel<<<512, 128, 0, stream>>>(featsbf, w1bf, w2bf, wdiff, idxb,
                                        scale1, shift1, scale2, shift2, out);
}
